// DragonFlySparsityPlugin_14998025798390
// MI455X (gfx1250) — matrix operand order repaired and checked by reading, not compiled, not run on hardware
//
#include <hip/hip_runtime.h>
#include <hip/hip_bf16.h>

typedef __attribute__((ext_vector_type(16))) _Float16 v16h;
typedef __attribute__((ext_vector_type(8)))  float    v8f;

#define D_DIM   1024
#define SEQ     2880
#define NLOW    576
#define PATCHES 24
#define GL      24      // low rows per patch
#define GH      96      // high rows per patch
#define TOPK    8
#define OUTROWS 768     // 576 + 24*8
#define BLOCK   384     // 12 waves of 32

__global__ __launch_bounds__(BLOCK)
void dragonfly_topk_kernel(const float* __restrict__ x, float* __restrict__ out) {
    const int bp   = blockIdx.x;          // 0..383  (b * 24 + p)
    const int b    = bp / PATCHES;
    const int p    = bp % PATCHES;
    const int tid  = threadIdx.x;
    const int w    = tid >> 5;            // wave 0..11
    const int lane = tid & 31;

    __shared__ float     qsum[D_DIM];     // un-normalized query accumulator (f32)
    __shared__ _Float16  qh[D_DIM];       // query in f16 for WMMA A
    __shared__ float     sspart[12][16];  // per-wave partial sum-of-squares (high rows)
    __shared__ float     inpart[12][16];  // per-wave partial inner products
    __shared__ float     innerv[GH];      // final scores
    __shared__ float     rnormv[GH];      // 1/||high_row|| (f32-exact)
    __shared__ int       topidx[TOPK];

    for (int i = tid; i < D_DIM; i += BLOCK) qsum[i] = 0.0f;
    __syncthreads();

    const float* xb = x   + (size_t)b * SEQ     * D_DIM;
    float*       ob = out + (size_t)b * OUTROWS * D_DIM;

    // ---------------- Phase 1: low rows (normalize -> out, accumulate q) ----
    #pragma unroll
    for (int rr = 0; rr < 2; ++rr) {
        const int rl = w * 2 + rr;                           // 0..23 within patch
        const float* row = xb + (size_t)(p * GL + rl) * D_DIM;
        float4 v[8];
        float  ss = 0.0f;
        #pragma unroll
        for (int t = 0; t < 8; ++t) {
            v[t] = ((const float4*)row)[t * 32 + lane];      // coalesced 512B/wave/iter
            ss += v[t].x * v[t].x + v[t].y * v[t].y + v[t].z * v[t].z + v[t].w * v[t].w;
        }
        #pragma unroll
        for (int off = 16; off >= 1; off >>= 1) ss += __shfl_xor(ss, off);
        const float rn = 1.0f / sqrtf(ss);
        float* orow = ob + (size_t)(p * GL + rl) * D_DIM;
        #pragma unroll
        for (int t = 0; t < 8; ++t) {
            float4 nv = make_float4(v[t].x * rn, v[t].y * rn, v[t].z * rn, v[t].w * rn);
            ((float4*)orow)[t * 32 + lane] = nv;
            const int base = (t * 32 + lane) * 4;
            atomicAdd(&qsum[base + 0], nv.x);
            atomicAdd(&qsum[base + 1], nv.y);
            atomicAdd(&qsum[base + 2], nv.z);
            atomicAdd(&qsum[base + 3], nv.w);
        }
    }
    __syncthreads();

    // ---------------- Phase 2: q -> f16 (skip /24: positive scale keeps order)
    for (int i = tid; i < D_DIM; i += BLOCK) qh[i] = (_Float16)qsum[i];
    __syncthreads();

    // ---------------- Phase 3: high rows: fused norm + WMMA scoring ---------
    {
        const int t6   = w % 6;           // 16-row tile
        const int kh   = w / 6;           // K half (0: c 0..15, 1: c 16..31)
        const int rloc = lane & 15;
        const int half = lane >> 4;       // B operand lane half -> K sub-block
        const int g    = t6 * 16 + rloc;
        const float* row = xb + (size_t)(NLOW + p * GH + g) * D_DIM;

        v8f   acc = {};
        float ss  = 0.0f;
        const int c0 = kh * 16;
        for (int c = c0; c < c0 + 16; ++c) {
            const int col0 = c * 32 + half * 16;
            float4 f[4];
            #pragma unroll
            for (int u = 0; u < 4; ++u) f[u] = ((const float4*)(row + col0))[u];

            v16h bv;
            #pragma unroll
            for (int u = 0; u < 4; ++u) {
                ss += f[u].x * f[u].x + f[u].y * f[u].y + f[u].z * f[u].z + f[u].w * f[u].w;
                bv[4 * u + 0] = (_Float16)f[u].x;
                bv[4 * u + 1] = (_Float16)f[u].y;
                bv[4 * u + 2] = (_Float16)f[u].z;
                bv[4 * u + 3] = (_Float16)f[u].w;
            }
            // A = q replicated over M rows; contiguous K order (k = 16*half + s), matching B
            v16h av;
            const int qbase = c * 32 + half * 16;
            #pragma unroll
            for (int h = 0; h < 8; ++h) av[h]     = qh[qbase + h];
            #pragma unroll
            for (int h = 0; h < 8; ++h) av[8 + h] = qh[qbase + 8 + h];

            acc = __builtin_amdgcn_wmma_f32_16x16x32_f16(
                      false, av, false, bv, (short)0, acc, false, false);
        }
        ss += __shfl_xor(ss, 16);         // combine the two column halves of the row
        if (lane < 16) {
            sspart[w][lane] = ss;
            inpart[w][lane] = acc[0];     // D VGPR0: N = lane -> this lane's own row
        }
    }
    __syncthreads();

    if (w < 6 && lane < 16) {
        const int g   = w * 16 + lane;
        const float s = sspart[w][lane] + sspart[w + 6][lane];
        const float r = 1.0f / sqrtf(s);
        rnormv[g] = r;
        innerv[g] = (inpart[w][lane] + inpart[w + 6][lane]) * r;
    }
    __syncthreads();

    // ---------------- Phase 4: top-8 (descending, tie -> smaller index) -----
    if (w == 0) {
        float v0 = innerv[lane], v1 = innerv[lane + 32], v2 = innerv[lane + 64];
        int   i0 = lane,         i1 = lane + 32,         i2 = lane + 64;
        #pragma unroll
        for (int j = 0; j < TOPK; ++j) {
            float bv = v0; int bi = i0;
            if (v1 > bv || (v1 == bv && i1 < bi)) { bv = v1; bi = i1; }
            if (v2 > bv || (v2 == bv && i2 < bi)) { bv = v2; bi = i2; }
            #pragma unroll
            for (int off = 16; off >= 1; off >>= 1) {
                const float ov = __shfl_xor(bv, off);
                const int   oi = __shfl_xor(bi, off);
                if (ov > bv || (ov == bv && oi < bi)) { bv = ov; bi = oi; }
            }
            if (lane == 0) topidx[j] = bi;
            if (bi == i0) v0 = -3.0e38f;
            if (bi == i1) v1 = -3.0e38f;
            if (bi == i2) v2 = -3.0e38f;
        }
    }
    __syncthreads();

    // ---------------- Phase 5: gather selected high rows (L2-hot re-read) ---
    for (int e = tid; e < TOPK * (D_DIM / 4); e += BLOCK) {
        const int j = e >> 8;             // D/4 == 256 float4 per row
        const int i = e & 255;
        const int g = topidx[j];
        const float rn = rnormv[g];
        const float4 s = ((const float4*)(xb + (size_t)(NLOW + p * GH + g) * D_DIM))[i];
        float4 dv = make_float4(s.x * rn, s.y * rn, s.z * rn, s.w * rn);
        ((float4*)(ob + (size_t)(NLOW + p * TOPK + j) * D_DIM))[i] = dv;
    }
}

extern "C" void kernel_launch(void* const* d_in, const int* in_sizes, int n_in,
                              void* d_out, int out_size, void* d_ws, size_t ws_size,
                              hipStream_t stream) {
    (void)in_sizes; (void)n_in; (void)d_ws; (void)ws_size; (void)out_size;
    const float* x = (const float*)d_in[0];
    float* out = (float*)d_out;
    dragonfly_topk_kernel<<<dim3(16 * PATCHES), dim3(BLOCK), 0, stream>>>(x, out);
}
